// RGLRU_82721070121000
// MI455X (gfx1250) — matrix operand order repaired and checked by reading, not compiled, not run on hardware
//
#include <hip/hip_runtime.h>
#include <hip/hip_bf16.h>
#include <cstdint>
#include <cstddef>

typedef __bf16 bf16_t;
typedef __attribute__((ext_vector_type(16))) __bf16 v16bf;
typedef __attribute__((ext_vector_type(8)))  __bf16 v8bf;
typedef __attribute__((ext_vector_type(4)))  __bf16 v4bf;
typedef __attribute__((ext_vector_type(8)))  float  v8f;
typedef __attribute__((ext_vector_type(4)))  float  v4f;
typedef __attribute__((ext_vector_type(4)))  int    v4i;

static constexpr float C_EXP = 8.0f;

#if defined(__AMDGCN__) && __has_builtin(__builtin_amdgcn_global_load_async_to_lds_b128) && \
    __has_builtin(__builtin_amdgcn_s_wait_asynccnt)
#define HAVE_ASYNC_LDS 1
#else
#define HAVE_ASYNC_LDS 0
#endif

#define AS1 __attribute__((address_space(1)))
#define AS3 __attribute__((address_space(3)))

#if HAVE_ASYNC_LDS
// global -> LDS async copy, 16B per lane, tracked by ASYNCcnt.
// Builtin signature (from clang diagnostic): (v4i AS1*, v4i AS3*, imm offset, imm cpol)
__device__ __forceinline__ void async_copy_b128(const void* g, void* l) {
  __builtin_amdgcn_global_load_async_to_lds_b128(
      (AS1 v4i*)(uintptr_t)g, (AS3 v4i*)(uintptr_t)l, 0, 0);
}
#endif

// ---------------- fp32 -> bf16 conversion (4 elements / thread) ----------------
__global__ __launch_bounds__(256)
void k_cvt_bf16_x4(const float* __restrict__ src, bf16_t* __restrict__ dst, long n4) {
  long i = (long)blockIdx.x * blockDim.x + threadIdx.x;
  if (i >= n4) return;
  v4f v = *(const v4f*)(src + 4 * i);
  v4bf o;
  o[0] = (bf16_t)v[0]; o[1] = (bf16_t)v[1]; o[2] = (bf16_t)v[2]; o[3] = (bf16_t)v[3];
  *(v4bf*)(dst + 4 * i) = o;
}

// ---------------- stable log_sigmoid(lambd) ----------------
__global__ __launch_bounds__(256)
void k_logsig(const float* __restrict__ lambd, float* __restrict__ out, int D) {
  int d = blockIdx.x * blockDim.x + threadIdx.x;
  if (d < D) {
    float l = lambd[d];
    out[d] = fminf(l, 0.0f) - log1pf(__expf(-fabsf(l)));
  }
}

// ---------------- WMMA fragment helpers ----------------
__device__ __forceinline__ v16bf make_frag(const bf16_t* p0, const bf16_t* p1) {
  v8bf lo = *(const v8bf*)p0;
  v8bf hi = *(const v8bf*)p1;
  v16bf f;
#pragma unroll
  for (int q = 0; q < 8; ++q) { f[q] = lo[q]; f[q + 8] = hi[q]; }
  return f;
}

// One BK=32 step of the dual GEMM for this wave (2x2 16x16 tiles, both W matrices).
// Tiles are row-major [rows][32] in LDS.
__device__ __forceinline__ void mma_step(const bf16_t* xsb, const bf16_t* wrsb,
                                         const bf16_t* wisb, int lane, int wm, int wn,
                                         v8f (&accr)[2][2], v8f (&acci)[2][2]) {
  // A: lane holds row m = lane&15; K chunks {ak..ak+7, ak+16..ak+23}
  const int am = lane & 15;
  const int ak = (lane < 16) ? 0 : 8;
  v16bf afrag[2];
#pragma unroll
  for (int mi = 0; mi < 2; ++mi) {
    const bf16_t* row = xsb + (wm * 32 + mi * 16 + am) * 32;
    afrag[mi] = make_frag(row + ak, row + ak + 16);
  }
  // B: lane holds col n = lane&15; K chunks {bk..bk+7, bk+16..bk+23}
  const int bn = lane & 15;
  const int bk = (lane < 16) ? 0 : 8;
#pragma unroll
  for (int ni = 0; ni < 2; ++ni) {
    const bf16_t* rrow = wrsb + (wn * 32 + ni * 16 + bn) * 32;
    const bf16_t* irow = wisb + (wn * 32 + ni * 16 + bn) * 32;
    v16bf brf = make_frag(rrow + bk, rrow + bk + 16);
    v16bf bif = make_frag(irow + bk, irow + bk + 16);
#pragma unroll
    for (int mi = 0; mi < 2; ++mi) {
      accr[mi][ni] = __builtin_amdgcn_wmma_f32_16x16x32_bf16(
          false, afrag[mi], false, brf, (short)0, accr[mi][ni], false, false);
      acci[mi][ni] = __builtin_amdgcn_wmma_f32_16x16x32_bf16(
          false, afrag[mi], false, bif, (short)0, acci[mi][ni], false, false);
    }
  }
}

// ---------------- fused dual-GEMM + gate epilogue ----------------
// r = sigmoid(x@W_r^T), i = sigmoid(x@W_i^T) via bf16 WMMA (f32 accum), then
// a = exp(8*r*logsig(lambd)) -> Aout, b = sqrt(max(1-a^2,0))*i*x -> Bout (d_out).
__global__ __launch_bounds__(256)
void k_gemm_gates(const bf16_t* __restrict__ Xb,    // [BT, D] bf16
                  const bf16_t* __restrict__ Wrb,   // [D, D]  bf16
                  const bf16_t* __restrict__ Wib,   // [D, D]  bf16
                  const float*  __restrict__ Xf,    // [BT, D] fp32 (original x)
                  const float*  __restrict__ logsig,// [D]
                  float* __restrict__ Aout,         // [BT, D]
                  float* __restrict__ Bout,         // [BT, D]
                  int D) {
  constexpr int BM = 128, BN = 64, BK = 32;

  const int tid  = threadIdx.x;
  const int lane = tid & 31;
  const int wid  = tid >> 5;   // 8 waves
  const int wm   = wid >> 1;   // 0..3 : 32-row strip
  const int wn   = wid & 1;    // 0..1 : 32-col strip

  const long m0 = (long)blockIdx.x * BM;
  const int  n0 = (int)blockIdx.y * BN;

  v8f accr[2][2], acci[2][2];
#pragma unroll
  for (int mi = 0; mi < 2; ++mi)
#pragma unroll
    for (int ni = 0; ni < 2; ++ni) { accr[mi][ni] = (v8f)(0.0f); acci[mi][ni] = (v8f)(0.0f); }

  const int lr = tid >> 2;        // 0..63  loader row
  const int lc = (tid & 3) * 8;   // 0,8,16,24 loader col (8 bf16 = 16B)
  const bf16_t* gx0 = Xb + (m0 + lr) * D + lc;
  const bf16_t* gx1 = Xb + (m0 + lr + 64) * D + lc;
  const bf16_t* gwr = Wrb + (long)(n0 + lr) * D + lc;
  const bf16_t* gwi = Wib + (long)(n0 + lr) * D + lc;

#if HAVE_ASYNC_LDS
  // Double-buffered pipeline on GLOBAL_LOAD_ASYNC_TO_LDS_B128 (ASYNCcnt tracked).
  __shared__ alignas(16) bf16_t xs [2][BM][BK];
  __shared__ alignas(16) bf16_t wrs[2][BN][BK];
  __shared__ alignas(16) bf16_t wis[2][BN][BK];

  auto issue = [&](int k0, int buf) {
    async_copy_b128(gx0 + k0, &xs[buf][lr][lc]);
    async_copy_b128(gx1 + k0, &xs[buf][lr + 64][lc]);
    async_copy_b128(gwr + k0, &wrs[buf][lr][lc]);
    async_copy_b128(gwi + k0, &wis[buf][lr][lc]);
  };

  issue(0, 0);                       // 4 async ops in flight
  int cur = 0;
  for (int k0 = 0; k0 < D; k0 += BK) {
    const bool has_next = (k0 + BK < D);
    if (has_next) {
      issue(k0 + BK, cur ^ 1);       // 8 in flight
      __builtin_amdgcn_s_wait_asynccnt(4);  // in-order done => current tile landed
    } else {
      __builtin_amdgcn_s_wait_asynccnt(0);
    }
    __syncthreads();                 // all waves' copies visible
    mma_step(&xs[cur][0][0], &wrs[cur][0][0], &wis[cur][0][0], lane, wm, wn, accr, acci);
    __syncthreads();                 // reads done before buffer is overwritten
    cur ^= 1;
  }
#else
  // Fallback: synchronous global->VGPR->LDS staging.
  __shared__ alignas(16) bf16_t xs [BM][BK];
  __shared__ alignas(16) bf16_t wrs[BN][BK];
  __shared__ alignas(16) bf16_t wis[BN][BK];

  for (int k0 = 0; k0 < D; k0 += BK) {
    __builtin_prefetch(gx0 + k0 + BK, 0, 3);   // WGP-scope prefetch of next tile
    __builtin_prefetch(gwr + k0 + BK, 0, 3);
    __builtin_prefetch(gwi + k0 + BK, 0, 3);
    __syncthreads();
    *(v8bf*)&xs [lr     ][lc] = *(const v8bf*)(gx0 + k0);
    *(v8bf*)&xs [lr + 64][lc] = *(const v8bf*)(gx1 + k0);
    *(v8bf*)&wrs[lr     ][lc] = *(const v8bf*)(gwr + k0);
    *(v8bf*)&wis[lr     ][lc] = *(const v8bf*)(gwi + k0);
    __syncthreads();
    mma_step(&xs[0][0], &wrs[0][0], &wis[0][0], lane, wm, wn, accr, acci);
  }
#endif

  // Epilogue. C/D layout: VGPR j -> (M = j or j+8, N = lane&15).
  const int cm = (lane < 16) ? 0 : 8;
  const int cn = lane & 15;
#pragma unroll
  for (int mi = 0; mi < 2; ++mi) {
#pragma unroll
    for (int ni = 0; ni < 2; ++ni) {
      const int colg = n0 + wn * 32 + ni * 16 + cn;
      const float ls = logsig[colg];
#pragma unroll
      for (int j = 0; j < 8; ++j) {
        const long rowg = m0 + wm * 32 + mi * 16 + cm + j;
        const size_t off = (size_t)rowg * D + colg;
        const float rg = 1.0f / (1.0f + __expf(-accr[mi][ni][j]));
        const float ig = 1.0f / (1.0f + __expf(-acci[mi][ni][j]));
        const float la = C_EXP * rg * ls;          // log a <= 0
        const float a  = __expf(la);
        const float om = fmaxf(1.0f - __expf(2.0f * la), 0.0f);
        const float bb = sqrtf(om) * ig * Xf[off];
        Aout[off] = a;
        Bout[off] = bb;
      }
    }
  }
}

// ---------------- chunked linear-recurrence scan over T ----------------
__global__ __launch_bounds__(256)
void k_scan_chunks(const float* __restrict__ A, const float* __restrict__ Bv,
                   float* __restrict__ Aagg, float* __restrict__ Bagg,
                   int T, int D, int CH) {
  const int d = blockIdx.x * blockDim.x + threadIdx.x;
  const int c = blockIdx.y;
  const int b = blockIdx.z;
  const int NC = gridDim.y;
  size_t base = ((size_t)b * T + (size_t)c * CH) * (size_t)D + d;
  float Ar = 1.0f, Br = 0.0f;
  for (int t = 0; t < CH; ++t) {
    const float a  = A[base];
    const float bb = Bv[base];
    Br = fmaf(a, Br, bb);
    Ar *= a;
    base += D;
  }
  const size_t o = ((size_t)b * NC + c) * (size_t)D + d;
  Aagg[o] = Ar;
  Bagg[o] = Br;
}

__global__ __launch_bounds__(256)
void k_scan_spine(const float* __restrict__ Aagg, float* __restrict__ Bagg,
                  int NC, int D) {
  const int idx = blockIdx.x * blockDim.x + threadIdx.x;   // over B*D
  const int b = idx / D;
  const int d = idx - b * D;
  float H = 0.0f;
  size_t o = (size_t)b * NC * D + d;
  for (int c = 0; c < NC; ++c) {
    const float a2 = Aagg[o];
    const float b2 = Bagg[o];
    Bagg[o] = H;              // h entering chunk c (exclusive scan)
    H = fmaf(a2, H, b2);
    o += D;
  }
}

__global__ __launch_bounds__(256)
void k_scan_apply(const float* __restrict__ A, const float* __restrict__ Hinit,
                  float* __restrict__ HB, int T, int D, int CH) {
  const int d = blockIdx.x * blockDim.x + threadIdx.x;
  const int c = blockIdx.y;
  const int b = blockIdx.z;
  const int NC = gridDim.y;
  float h = Hinit[((size_t)b * NC + c) * (size_t)D + d];
  size_t base = ((size_t)b * T + (size_t)c * CH) * (size_t)D + d;
  for (int t = 0; t < CH; ++t) {
    const float a  = A[base];
    const float bb = HB[base];
    h = fmaf(a, h, bb);
    HB[base] = h;
    base += D;
  }
}

extern "C" void kernel_launch(void* const* d_in, const int* in_sizes, int n_in,
                              void* d_out, int out_size, void* d_ws, size_t ws_size,
                              hipStream_t stream) {
  const float* x     = (const float*)d_in[0];
  const float* W_r   = (const float*)d_in[1];
  const float* W_i   = (const float*)d_in[2];
  const float* lambd = (const float*)d_in[3];

  const int  D   = in_sizes[3];                  // 1024
  const long BT  = (long)in_sizes[0] / D;        // 32768
  const int  Bsz = 8;                            // reference setup: B=8
  const int  T   = (int)(BT / Bsz);              // 4096
  const int  CH  = 64;
  const int  NC  = T / CH;                       // 64

  uint8_t* w = (uint8_t*)d_ws;
  size_t off = 0;
  auto alloc = [&](size_t bytes) -> void* {
    void* p = w + off;
    off = (off + bytes + 255) & ~(size_t)255;
    return p;
  };
  bf16_t* xb   = (bf16_t*)alloc((size_t)BT * D * sizeof(bf16_t));       // 64 MiB
  bf16_t* wrb  = (bf16_t*)alloc((size_t)D * D * sizeof(bf16_t));        //  2 MiB
  bf16_t* wib  = (bf16_t*)alloc((size_t)D * D * sizeof(bf16_t));        //  2 MiB
  float*  lsg  = (float*) alloc((size_t)D * sizeof(float));
  float*  Aarr = (float*) alloc((size_t)BT * D * sizeof(float));        // 128 MiB
  float*  Aagg = (float*) alloc((size_t)Bsz * NC * D * sizeof(float));  //  2 MiB
  float*  Bagg = (float*) alloc((size_t)Bsz * NC * D * sizeof(float));  //  2 MiB
  float*  Bout = (float*)d_out;  // holds b, then overwritten with h

  {
    const long n4 = (BT * (long)D) / 4;
    k_cvt_bf16_x4<<<(unsigned)((n4 + 255) / 256), 256, 0, stream>>>(x, xb, n4);
    const long w4 = ((long)D * D) / 4;
    k_cvt_bf16_x4<<<(unsigned)((w4 + 255) / 256), 256, 0, stream>>>(W_r, wrb, w4);
    k_cvt_bf16_x4<<<(unsigned)((w4 + 255) / 256), 256, 0, stream>>>(W_i, wib, w4);
    k_logsig<<<(D + 255) / 256, 256, 0, stream>>>(lambd, lsg, D);
  }

  dim3 gg((unsigned)(BT / 128), (unsigned)(D / 64));
  k_gemm_gates<<<gg, 256, 0, stream>>>(xb, wrb, wib, x, lsg, Aarr, Bout, D);

  dim3 gs((unsigned)(D / 256), (unsigned)NC, (unsigned)Bsz);
  k_scan_chunks<<<gs, 256, 0, stream>>>(Aarr, Bout, Aagg, Bagg, T, D, CH);
  k_scan_spine<<<(unsigned)((Bsz * D) / 256), 256, 0, stream>>>(Aagg, Bagg, NC, D);
  k_scan_apply<<<gs, 256, 0, stream>>>(Aarr, Bagg, Bout, T, D, CH);
}
